// EndToEndBertBrain_84310208020704
// MI455X (gfx1250) — matrix operand order repaired and checked by reading, not compiled, not run on hardware
//
#include <hip/hip_runtime.h>

typedef __attribute__((ext_vector_type(16))) __bf16 v16bf;
typedef __attribute__((ext_vector_type(8)))  __bf16 v8bf;
typedef __attribute__((ext_vector_type(8)))  float  v8f;

#define NS   4
#define NSRC 2000
#define DIM  1024
#define NTR  300
#define TRIM 10
#define TT   280            // trimmed rows per story
#define MROW (NS*TT)        // 1120
#define KD   (DIM*4)        // 4096
#define NV   50000

#define BM 128
#define BN 128
#define BK 32
#define NKT (KD / BK)       // 128 K-steps
#define LDA 40   // bf16 units per A row in LDS (padded, 80B = 16B-aligned rows)
#define LDB 40   // bf16 units per B column in LDS (padded)

__device__ __forceinline__ v8f fzero8() {
    v8f z;
#pragma unroll
    for (int i = 0; i < 8; ++i) z[i] = 0.0f;
    return z;
}

// Single-instruction packed conversion: dst[15:0]=bf16(a), dst[31:16]=bf16(b).
__device__ __forceinline__ unsigned int pack_bf16x2(float a, float b) {
    unsigned int r;
    asm("v_cvt_pk_bf16_f32 %0, %1, %2" : "=v"(r) : "v"(a), "v"(b));
    return r;
}

// Async copy of 32 contiguous bytes global -> LDS (2x b128, ASYNCcnt-tracked).
// Same INST_OFFSET applies to both the LDS and global address (ISA 08 §4.4).
__device__ __forceinline__ void async_copy32(unsigned lds_dst, const __bf16* gptr) {
    unsigned long long ga = (unsigned long long)(uintptr_t)gptr;
    asm volatile("global_load_async_to_lds_b128 %0, %1, off\n\t"
                 "global_load_async_to_lds_b128 %0, %1, off offset:16"
                 :: "v"(lds_dst), "v"(ga) : "memory");
}
__device__ __forceinline__ void wait_async0() {
    asm volatile("s_wait_asynccnt 0x0" ::: "memory");
}

// ---------------------------------------------------------------------------
// Stage 1: Lanczos downsample (fused row-normalize), trimmed rows only.
// ---------------------------------------------------------------------------
__global__ __launch_bounds__(256) void lanczos_kernel(
    const float* __restrict__ emb, const float* __restrict__ data_t,
    const float* __restrict__ tr_t, float* __restrict__ feat_raw)
{
    const int s    = blockIdx.y;
    const int trow = blockIdx.x;       // 0..279
    const int i    = trow + TRIM;
    const int tid  = threadIdx.x;

    __shared__ float wl[NSRC];
    __shared__ float red[256];
    __shared__ int   jmn_s, jmx_s;

    const float* dts = data_t + s * NSRC;
    const float* trs = tr_t   + s * NTR;
    const float scale = (dts[NSRC - 1] - dts[0]) / (trs[NTR - 1] - trs[0]);
    const float ti    = trs[i];

    if (tid == 0) { jmn_s = NSRC; jmx_s = -1; }
    __syncthreads();

    const float PI = 3.14159265358979f;
    float lsum = 0.0f;
    int lmn = NSRC, lmx = -1;
    for (int j = tid; j < NSRC; j += 256) {
        float x  = (ti - dts[j]) * scale;
        float ax = fabsf(x);
        float w  = 0.0f;
        if (ax < 3.0f) {
            if (ax < 1e-8f) w = 1.0f;
            else            w = 3.0f * __sinf(PI * x) * __sinf(PI * x * (1.0f / 3.0f))
                                / (PI * PI * x * x);
            lmn = min(lmn, j); lmx = max(lmx, j);
        }
        wl[j] = w;
        lsum += w;
    }
    red[tid] = lsum;
    atomicMin(&jmn_s, lmn);
    atomicMax(&jmx_s, lmx);
    __syncthreads();
#pragma unroll
    for (int off = 128; off > 0; off >>= 1) {
        if (tid < off) red[tid] += red[tid + off];
        __syncthreads();
    }
    const float ssum = red[0];
    const float invs = (ssum > 0.0f) ? (1.0f / ssum) : 1.0f;
    const int j0 = jmn_s, j1 = jmx_s;

    float ax0 = 0.f, ax1 = 0.f, ax2 = 0.f, ax3 = 0.f;
    const float4* e4 = (const float4*)(emb + (size_t)s * NSRC * DIM);
    for (int j = j0; j <= j1; ++j) {
        float w = wl[j];
        if (w != 0.0f) {
            float4 e = e4[(size_t)j * (DIM / 4) + tid];
            ax0 = fmaf(w, e.x, ax0); ax1 = fmaf(w, e.y, ax1);
            ax2 = fmaf(w, e.z, ax2); ax3 = fmaf(w, e.w, ax3);
        }
    }
    float4 o; o.x = ax0 * invs; o.y = ax1 * invs; o.z = ax2 * invs; o.w = ax3 * invs;
    float4* outp = (float4*)(feat_raw + ((size_t)(s * TT + trow)) * DIM);
    outp[tid] = o;
}

// ---------------------------------------------------------------------------
// Stage 2: per-(story,column) z-score over 280 rows, ddof=1, +EPS.
// ---------------------------------------------------------------------------
__global__ __launch_bounds__(256) void norm_kernel(
    const float* __restrict__ feat_raw, float* __restrict__ feat)
{
    const int gid = blockIdx.x * 256 + threadIdx.x;   // 0..4095
    const int s = gid >> 10, c = gid & (DIM - 1);
    const float* p = feat_raw + (size_t)s * TT * DIM + c;
    float sum = 0.f, sq = 0.f;
    for (int t = 0; t < TT; ++t) {
        float v = p[(size_t)t * DIM];
        sum += v; sq = fmaf(v, v, sq);
    }
    const float mean = sum * (1.0f / TT);
    const float var  = (sq - sum * mean) * (1.0f / (TT - 1));
    const float inv  = 1.0f / (sqrtf(fmaxf(var, 0.0f)) + 1e-6f);
    float* o = feat + (size_t)s * TT * DIM + c;
    for (int t = 0; t < TT; ++t)
        o[(size_t)t * DIM] = (p[(size_t)t * DIM] - mean) * inv;
}

// ---------------------------------------------------------------------------
// Stage 3: build delayed matrix (f32 output #2) and bf16 copy for WMMA GEMM.
// ---------------------------------------------------------------------------
__global__ __launch_bounds__(256) void delay_kernel(
    const float* __restrict__ feat, float* __restrict__ dly,
    __bf16* __restrict__ abf)
{
    const size_t gid = (size_t)blockIdx.x * 256 + threadIdx.x;
    if (gid >= (size_t)MROW * KD) return;
    const int r   = (int)(gid >> 12);
    const int col = (int)(gid & (KD - 1));
    const int dd  = col >> 10;
    const int c   = col & (DIM - 1);
    const float v = (r >= dd) ? feat[(size_t)(r - dd) * DIM + c] : 0.0f;
    dly[gid] = v;
    abf[gid] = (__bf16)v;
}

// ---------------------------------------------------------------------------
// Stage 4: C(1120x50000) = A_bf16(1120x4096) x W(4096x50000) + b
// 128x128x32 tiles, double-buffered LDS.  A tile staged via
// global_load_async_to_lds_b128 (ASYNCcnt); W tile loaded to VGPRs,
// v_cvt_pk_bf16_f32 and stored transposed [n][k].  8 waves, 8 WMMA each.
// ---------------------------------------------------------------------------
__global__ __launch_bounds__(256) void gemm_kernel(
    const __bf16* __restrict__ A, const float* __restrict__ W,
    const float* __restrict__ bias, float* __restrict__ out)
{
    __shared__ __bf16 sa[2][BM * LDA];
    __shared__ __bf16 sb[2][BN * LDB];

    const int tid  = threadIdx.x;
    const int m0   = blockIdx.x * BM;
    const int n0   = blockIdx.y * BN;
    const int wave = tid >> 5, lane = tid & 31;
    const int wm = wave & 3;          // 4 m-groups of 32 rows
    const int wn = wave >> 2;         // 2 n-groups of 64 cols
    const int khalf = lane >> 4, l16 = lane & 15;

    v8f acc[2][4];
#pragma unroll
    for (int mt = 0; mt < 2; ++mt)
#pragma unroll
        for (int nt = 0; nt < 4; ++nt) acc[mt][nt] = fzero8();

    // ---- A staging: thread -> (row, 16-bf16 half-row); clamped (branchless)
    const int ar = tid >> 1;
    const int ak = (tid & 1) * 16;
    const int arow = min(m0 + ar, MROW - 1);
    const __bf16* Abase = A + (size_t)arow * KD + ak;
    const unsigned ldsA[2] = {
        (unsigned)(uintptr_t)(&sa[0][ar * LDA + ak]),
        (unsigned)(uintptr_t)(&sa[1][ar * LDA + ak]) };

    // ---- W staging: thread -> (4 k-rows, 4 n-cols); clamped (branchless)
    const int bn = (tid & 31) * 4;
    const int bk = (tid >> 5) * 4;
    const int ncol = min(n0 + bn, NV - 4);

    float4 wv[4];

    // ---------------- prologue: tile 0 into buffer 0 ----------------
    async_copy32(ldsA[0], Abase);
#pragma unroll
    for (int p = 0; p < 4; ++p)
        wv[p] = *(const float4*)(W + (size_t)(bk + p) * NV + ncol);
    {
        const float* f0 = (const float*)&wv[0];
        const float* f1 = (const float*)&wv[1];
        const float* f2 = (const float*)&wv[2];
        const float* f3 = (const float*)&wv[3];
#pragma unroll
        for (int j = 0; j < 4; ++j) {
            unsigned int* q = (unsigned int*)(&sb[0][(bn + j) * LDB + bk]);
            q[0] = pack_bf16x2(f0[j], f1[j]);
            q[1] = pack_bf16x2(f2[j], f3[j]);
        }
    }
    wait_async0();
    __syncthreads();

    // ---------------- main loop ----------------
    for (int i = 0; i < NKT; ++i) {
        const int cur = i & 1, nxt = cur ^ 1;
        const bool more = (i + 1) < NKT;

        if (more) {
            const int kt1 = (i + 1) * BK;
            async_copy32(ldsA[nxt], Abase + kt1);
#pragma unroll
            for (int p = 0; p < 4; ++p)
                wv[p] = *(const float4*)(W + (size_t)(kt1 + bk + p) * NV + ncol);
            if (i + 2 < NKT)   // -> global_prefetch_b8 for the tile after next
                __builtin_prefetch(W + (size_t)((i + 2) * BK + bk) * NV + ncol, 0, 1);
        }

        // ---- compute from buffer `cur` (documented wave32 WMMA layouts)
        const __bf16* pa = sa[cur];
        const __bf16* pb = sb[cur];
        v16bf afrag[2], bfrag[4];
#pragma unroll
        for (int mt = 0; mt < 2; ++mt) {
            const int row  = wm * 32 + mt * 16 + l16;
            const int base = row * LDA + khalf * 8;
            v8bf lo = *(const v8bf*)(pa + base);
            v8bf hi = *(const v8bf*)(pa + base + 16);
            afrag[mt] = __builtin_shufflevector(lo, hi, 0, 1, 2, 3, 4, 5, 6, 7,
                                                8, 9, 10, 11, 12, 13, 14, 15);
        }
#pragma unroll
        for (int nt = 0; nt < 4; ++nt) {
            const int coln = wn * 64 + nt * 16 + l16;
            const int base = coln * LDB + khalf * 8;
            v8bf lo = *(const v8bf*)(pb + base);
            v8bf hi = *(const v8bf*)(pb + base + 16);
            bfrag[nt] = __builtin_shufflevector(lo, hi, 0, 1, 2, 3, 4, 5, 6, 7,
                                                8, 9, 10, 11, 12, 13, 14, 15);
        }
#pragma unroll
        for (int mt = 0; mt < 2; ++mt)
#pragma unroll
            for (int nt = 0; nt < 4; ++nt)
                acc[mt][nt] = __builtin_amdgcn_wmma_f32_16x16x32_bf16(
                    false, afrag[mt], false, bfrag[nt],
                    (short)0, acc[mt][nt], false, false);

        if (more) {
            __syncthreads();    // everyone done reading buf `cur` (and earlier `nxt`)
            const float* f0 = (const float*)&wv[0];
            const float* f1 = (const float*)&wv[1];
            const float* f2 = (const float*)&wv[2];
            const float* f3 = (const float*)&wv[3];
#pragma unroll
            for (int j = 0; j < 4; ++j) {
                unsigned int* q = (unsigned int*)(&sb[nxt][(bn + j) * LDB + bk]);
                q[0] = pack_bf16x2(f0[j], f1[j]);
                q[1] = pack_bf16x2(f2[j], f3[j]);
            }
            wait_async0();      // this wave's async A(i+1) landed in LDS
            __syncthreads();    // all waves' B stores + async copies visible
        }
    }

    // ---- epilogue: VGPR v -> M = v + 8*khalf, N = l16 (32-bit C/D layout)
#pragma unroll
    for (int nt = 0; nt < 4; ++nt) {
        const int coln = n0 + wn * 64 + nt * 16 + l16;
        if (coln < NV) {
            const float bb = bias[coln];
#pragma unroll
            for (int mt = 0; mt < 2; ++mt) {
                const int rbase = m0 + wm * 32 + mt * 16 + 8 * khalf;
#pragma unroll
                for (int v = 0; v < 8; ++v) {
                    const int row = rbase + v;
                    if (row < MROW)
                        out[(size_t)row * NV + coln] = acc[mt][nt][v] + bb;
                }
            }
        }
    }
}

// ---------------------------------------------------------------------------
extern "C" void kernel_launch(void* const* d_in, const int* in_sizes, int n_in,
                              void* d_out, int out_size, void* d_ws, size_t ws_size,
                              hipStream_t stream) {
    const float* emb    = (const float*)d_in[0];   // 4 x 2000 x 1024
    const float* data_t = (const float*)d_in[1];   // 4 x 2000
    const float* tr_t   = (const float*)d_in[2];   // 4 x 300
    const float* W      = (const float*)d_in[3];   // 4096 x 50000
    const float* bias   = (const float*)d_in[4];   // 50000

    float* out_pred = (float*)d_out;                           // 1120 x 50000
    float* out_dly  = out_pred + (size_t)MROW * NV;            // 1120 x 4096

    char*  ws       = (char*)d_ws;
    float* feat_raw = (float*)ws;                               // 1120x1024 f32
    float* feat     = feat_raw + (size_t)MROW * DIM;            // 1120x1024 f32
    __bf16* abf     = (__bf16*)(ws + 2 * (size_t)MROW * DIM * sizeof(float));

    lanczos_kernel<<<dim3(TT, NS), 256, 0, stream>>>(emb, data_t, tr_t, feat_raw);
    norm_kernel<<<(NS * DIM) / 256, 256, 0, stream>>>(feat_raw, feat);
    delay_kernel<<<((size_t)MROW * KD + 255) / 256, 256, 0, stream>>>(feat, out_dly, abf);
    gemm_kernel<<<dim3((MROW + BM - 1) / BM, (NV + BN - 1) / BN), 256, 0, stream>>>(
        abf, W, bias, out_pred);
}
